// TrunkH_65807488910115
// MI455X (gfx1250) — matrix operand order repaired and checked by reading, not compiled, not run on hardware
//
#include <hip/hip_runtime.h>
#include <hip/hip_bf16.h>
#include <stddef.h>

// ---------------------------------------------------------------------------
// Problem constants (from reference)
// ---------------------------------------------------------------------------
#define BATCH 32
#define HH 28
#define WW 28
#define C1 64
#define DMODEL 256
#define NHEAD 8
#define HDIM 32
#define SEQ 784            // H*W
#define SEQP 800           // padded to multiple of 32 for K-dim of P@V
#define KSZ 9
#define KTH 803            // ceil(0.0005 * 32*64*28*28)
#define NCONV (BATCH*C1*HH*WW)   // 1,605,632
#define NPIX (BATCH*SEQ)         // 25,088

// ---------------------------------------------------------------------------
// CDNA5 WMMA types / helpers (wave32, 16x16x32 bf16 -> f32)
// ---------------------------------------------------------------------------
typedef __attribute__((ext_vector_type(16))) __bf16 v16bf;
typedef __attribute__((ext_vector_type(8)))  float  v8f;

__device__ __forceinline__ __bf16 f2bf(float f) {
    unsigned u = __float_as_uint(f);
    u += 0x7FFFu + ((u >> 16) & 1u);            // round-to-nearest-even
    unsigned short h = (unsigned short)(u >> 16);
    __bf16 r;
    __builtin_memcpy(&r, &h, 2);
    return r;
}

__device__ __forceinline__ v8f wmma_bf16(v16bf a, v16bf b, v8f c) {
    // emits v_wmma_f32_16x16x32_bf16
    return __builtin_amdgcn_wmma_f32_16x16x32_bf16(
        /*neg_a=*/false, a, /*neg_b=*/false, b,
        /*c_mod=*/(short)0, c, /*reuse_a=*/false, /*reuse_b=*/false);
}

// A fragment: 16x32 (MxK) bf16, row-major source, lda elements.
// Layout used here (not the ISA table k order): lane r=lane&15 holds row m0+r; hi=lane>>4 selects K half:
// contiguous 16 elements [hi*16, hi*16+16).
__device__ __forceinline__ v16bf load_a_frag(const __bf16* __restrict__ A,
                                             int lda, int m0, int kc, int lane) {
    int r = lane & 15, hi = lane >> 4;
    const __bf16* p = A + (size_t)(m0 + r) * lda + kc + hi * 16;
    v16bf a;
#pragma unroll
    for (int i = 0; i < 16; ++i) a[i] = p[i];
    return a;
}

// B fragment: 32x16 (KxN) where the math-B is Bw^T and Bw is row-major (N rows
// of K elems, ldb).  Lane n=lane&15 holds column n; hi selects K half:
// contiguous 16 elements Bw[n0+n][kc + hi*16 .. +15].
__device__ __forceinline__ v16bf load_b_frag_rowk(const __bf16* __restrict__ Bw,
                                                  int ldb, int n0, int kc, int lane) {
    int n = lane & 15, hi = lane >> 4;
    const __bf16* p = Bw + (size_t)(n0 + n) * ldb + kc + hi * 16;
    v16bf b;
#pragma unroll
    for (int i = 0; i < 16; ++i) b[i] = p[i];
    return b;
}

// B fragment from a KxN row-major matrix (true column gather, stride ldv).
__device__ __forceinline__ v16bf load_b_frag_strided(const __bf16* __restrict__ V,
                                                     int ldv, int k0, int n0, int lane) {
    int n = lane & 15, hi = lane >> 4;
    const __bf16* p = V + (size_t)(k0 + hi * 16) * ldv + n0 + n;
    v16bf b;
#pragma unroll
    for (int i = 0; i < 16; ++i) b[i] = p[(size_t)i * ldv];
    return b;
}

// ---------------------------------------------------------------------------
// 0) workspace init: zero padded QKV buffers, histograms, radix state
// ---------------------------------------------------------------------------
__global__ void init_ws_kernel(unsigned* __restrict__ qkv_words, size_t nwords,
                               unsigned* __restrict__ hist, unsigned* __restrict__ state) {
    size_t i = (size_t)blockIdx.x * blockDim.x + threadIdx.x;
    size_t stride = (size_t)gridDim.x * blockDim.x;
    for (size_t j = i; j < nwords; j += stride) qkv_words[j] = 0u;
    if (i < 1024) hist[i] = 0u;
    if (i == 0) { state[0] = 0u; state[1] = KTH; state[2] = 0u; }
}

// ---------------------------------------------------------------------------
// 1) 9x9 same-pad conv (1->64 ch) + ReLU
// ---------------------------------------------------------------------------
__global__ void conv_relu_kernel(const float* __restrict__ xin,
                                 const float* __restrict__ wconv,
                                 float* __restrict__ A) {
    int idx = blockIdx.x * blockDim.x + threadIdx.x;
    if (idx >= NCONV) return;
    int x = idx % WW;
    int y = (idx / WW) % HH;
    int c = (idx / (HH * WW)) % C1;
    int b = idx / (HH * WW * C1);
    const float* xb = xin + (size_t)b * HH * WW;
    const float* wc = wconv + c * (KSZ * KSZ);
    float s = 0.0f;
#pragma unroll
    for (int dy = 0; dy < KSZ; ++dy) {
        int iy = y + dy - KSZ / 2;
        if (iy < 0 || iy >= HH) continue;
#pragma unroll
        for (int dx = 0; dx < KSZ; ++dx) {
            int ix = x + dx - KSZ / 2;
            if (ix < 0 || ix >= WW) continue;
            s += xb[iy * WW + ix] * wc[dy * KSZ + dx];
        }
    }
    A[idx] = fmaxf(s, 0.0f);
}

// ---------------------------------------------------------------------------
// 2) exact k-th largest via 4-pass radix select on float bits (values >= 0)
// ---------------------------------------------------------------------------
__global__ void radix_hist_kernel(const float* __restrict__ A, int n,
                                  unsigned* __restrict__ hist,
                                  const unsigned* __restrict__ state, int p) {
    int shift = 24 - 8 * p;
    unsigned prefix = state[0];
    for (int i = blockIdx.x * blockDim.x + threadIdx.x; i < n;
         i += gridDim.x * blockDim.x) {
        unsigned u = __float_as_uint(A[i]);
        bool ok = (p == 0) || ((u >> (shift + 8)) == (prefix >> (shift + 8)));
        if (ok) atomicAdd(&hist[p * 256 + ((u >> shift) & 255u)], 1u);
    }
}

__global__ void radix_select_kernel(unsigned* __restrict__ state,
                                    const unsigned* __restrict__ hist, int p) {
    if (threadIdx.x != 0 || blockIdx.x != 0) return;
    int shift = 24 - 8 * p;
    unsigned kth = state[1];
    unsigned cum = 0;
    int sel = 0;
    for (int bin = 255; bin >= 0; --bin) {
        unsigned c = hist[p * 256 + bin];
        if (cum + c >= kth) { sel = bin; break; }
        cum += c;
    }
    state[0] |= ((unsigned)sel << shift);
    state[1] = kth - cum;
    if (p == 3) state[2] = state[0];
}

// ---------------------------------------------------------------------------
// 3) per-pixel masked top-2 sparse softmax; emits sparse_weights + top2_idx
//    to d_out and compact (w0,w1,c0,c1) for the embedding stage.
// ---------------------------------------------------------------------------
__global__ void pixel_select_kernel(const float* __restrict__ A,
                                    const unsigned* __restrict__ state,
                                    float* __restrict__ sw_out,
                                    float* __restrict__ t2_out,
                                    float* __restrict__ pixW,
                                    int* __restrict__ pixC) {
    int pix = blockIdx.x * blockDim.x + threadIdx.x;
    if (pix >= NPIX) return;
    int b = pix / SEQ, s = pix % SEQ;
    int y = s / WW, x = s % WW;
    float thr = __uint_as_float(state[2]);
    float v0 = -1.0e30f, v1 = -1.0e30f;
    int i0 = 0, i1 = 1, count = 0;
    for (int c = 0; c < C1; ++c) {
        float a = A[(((size_t)b * C1 + c) * HH + y) * WW + x];
        float vv = (a >= thr) ? a : 0.0f;
        if (vv > 0.0f) ++count;
        if (vv > v0) { v1 = v0; i1 = i0; v0 = vv; i0 = c; }
        else if (vv > v1) { v1 = vv; i1 = c; }
    }
    float w0 = 0.0f, w1 = 0.0f;
    if (count >= 2) {
        float e1 = __expf(v1 - v0);
        float inv = 1.0f / (1.0f + e1);
        w0 = inv; w1 = e1 * inv;
    } else if (count == 1) {
        w0 = 1.0f;
    }
    float* sw = sw_out + (size_t)pix * C1;
    for (int c = 0; c < C1; ++c) sw[c] = 0.0f;
    if (count >= 1) sw[i0] = w0;
    if (count >= 2) sw[i1] = w1;
    t2_out[pix * 2 + 0] = (float)i0;
    t2_out[pix * 2 + 1] = (float)i1;
    pixW[pix * 2 + 0] = w0; pixW[pix * 2 + 1] = w1;
    pixC[pix * 2 + 0] = i0; pixC[pix * 2 + 1] = i1;
}

// ---------------------------------------------------------------------------
// 4) sparse embed (<=2 nonzeros) + sinusoidal 2D pos-encoding -> seq (bf16)
// ---------------------------------------------------------------------------
__global__ void embed_kernel(const float* __restrict__ CE,
                             const float* __restrict__ pixW,
                             const int* __restrict__ pixC,
                             __bf16* __restrict__ seq) {
    int idx = blockIdx.x * blockDim.x + threadIdx.x;  // NPIX * DMODEL
    int d = idx & (DMODEL - 1);
    int pix = idx >> 8;
    if (pix >= NPIX) return;
    int s = pix % SEQ;
    int y = s / WW, x = s % WW;
    float w0 = pixW[pix * 2], w1 = pixW[pix * 2 + 1];
    int c0 = pixC[pix * 2], c1 = pixC[pix * 2 + 1];
    float e = w0 * CE[c0 * DMODEL + d] + w1 * CE[c1 * DMODEL + d];
    float div = __expf((float)(d & ~1) * (-9.210340371976184f / (float)DMODEL));
    float pe = (d & 1) ? __cosf((float)x * div) : __sinf((float)y * div);
    seq[idx] = f2bf(e + pe);
}

// ---------------------------------------------------------------------------
// 5) weight conversion f32 -> bf16 (in_proj 768x256, out_proj 256x256)
// ---------------------------------------------------------------------------
__global__ void convert_w_kernel(const float* __restrict__ wq,
                                 const float* __restrict__ wo,
                                 __bf16* __restrict__ wq_bf,
                                 __bf16* __restrict__ wo_bf) {
    int idx = blockIdx.x * blockDim.x + threadIdx.x;
    if (idx < 3 * DMODEL * DMODEL) wq_bf[idx] = f2bf(wq[idx]);
    if (idx < DMODEL * DMODEL)     wo_bf[idx] = f2bf(wo[idx]);
}

// ---------------------------------------------------------------------------
// 6) QKV projection GEMM: per-batch (784x256)@(256x768)^T + bias, WMMA bf16.
//    One 16x16 output tile per wave; q pre-scaled by 1/sqrt(hd); results
//    scattered into padded [b][head][SEQP][32] bf16 buffers.
// ---------------------------------------------------------------------------
__global__ void qkv_gemm_kernel(const __bf16* __restrict__ seq,
                                const __bf16* __restrict__ wqkv,
                                const float* __restrict__ bias,
                                __bf16* __restrict__ q,
                                __bf16* __restrict__ k,
                                __bf16* __restrict__ v) {
    int lane = threadIdx.x & 31, wave = threadIdx.x >> 5;
    int tile = blockIdx.x * 8 + wave;
    if (tile >= BATCH * 49 * 48) return;
    int b  = tile / (49 * 48);
    int t  = tile % (49 * 48);
    int m0 = (t / 48) * 16;
    int n0 = (t % 48) * 16;
    const __bf16* Aseq = seq + (size_t)b * SEQ * DMODEL;
    v8f acc = {};
#pragma unroll
    for (int kc = 0; kc < DMODEL; kc += 32) {
        __builtin_prefetch(Aseq + (size_t)m0 * DMODEL + kc + 32, 0, 0);
        v16bf a = load_a_frag(Aseq, DMODEL, m0, kc, lane);
        v16bf w = load_b_frag_rowk(wqkv, DMODEL, n0, kc, lane);
        acc = wmma_bf16(a, w, acc);
    }
    int r = lane & 15, hi = lane >> 4;
    int o = n0 + r;
    int which = o >> 8;            // 0=q 1=k 2=v
    int oo = o & (DMODEL - 1);
    int head = oo >> 5, col = oo & 31;
    __bf16* dst = (which == 0) ? q : (which == 1) ? k : v;
    float bs = bias[o];
    float scale = (which == 0) ? 0.17677669529663687f : 1.0f;  // 1/sqrt(32)
    size_t base = ((size_t)(b * NHEAD + head) * SEQP) * HDIM + col;
#pragma unroll
    for (int i = 0; i < 8; ++i) {
        int row = m0 + i + hi * 8;
        dst[base + (size_t)row * HDIM] = f2bf((acc[i] + bs) * scale);
    }
}

// ---------------------------------------------------------------------------
// 7) fused attention: per (b, head, 16-row q block):
//    scores (WMMA, K=hd=32 in one op) -> LDS f32 -> row softmax ->
//    bf16 probs in LDS -> P@V (WMMA over 25 K-chunks) -> ctx bf16.
//    128 threads = 4 waves; dynamic LDS ~77 KB (<< 320 KB/WGP).
// ---------------------------------------------------------------------------
__global__ void attn_kernel(const __bf16* __restrict__ q,
                            const __bf16* __restrict__ k,
                            const __bf16* __restrict__ v,
                            __bf16* __restrict__ ctx) {
    int blk = blockIdx.x;          // (b*8 + h)*49 + qt
    int qt = blk % 49;
    int bh = blk / 49;
    int b = bh >> 3, h = bh & 7;
    int tid = threadIdx.x, lane = tid & 31, wave = tid >> 5;

    extern __shared__ char smem[];
    float* scores  = (float*)smem;                   // 16 x SEQP
    float* rowm    = scores + 16 * SEQP;             // 16
    float* rowinv  = rowm + 16;                      // 16
    float* red     = rowinv + 16;                    // 2 x 16 x 16
    __bf16* probs  = (__bf16*)(red + 2 * 256);       // 16 x SEQP

    const __bf16* qh = q + (size_t)bh * SEQP * HDIM;
    const __bf16* kh = k + (size_t)bh * SEQP * HDIM;
    const __bf16* vh = v + (size_t)bh * SEQP * HDIM;
    int m0 = qt * 16;
    int r = lane & 15, hi = lane >> 4;

    // --- scores: Q(16x32) @ K^T(32x16) per key tile, one WMMA each --------
    v16bf aq = load_a_frag(qh, HDIM, m0, 0, lane);
    for (int kt = wave; kt < SEQP / 16; kt += 4) {
        v16bf bk = load_b_frag_rowk(kh, HDIM, kt * 16, 0, lane);
        v8f c = {};
        c = wmma_bf16(aq, bk, c);
        int ncol = kt * 16 + r;
#pragma unroll
        for (int i = 0; i < 8; ++i) {
            int row = i + hi * 8;
            scores[row * SEQP + ncol] = (ncol < SEQ) ? c[i] : -1.0e30f;
        }
    }
    __syncthreads();

    // --- row softmax stats ------------------------------------------------
    if (tid < 16) {
        float m = -1.0e30f;
        for (int c = 0; c < SEQP; ++c) m = fmaxf(m, scores[tid * SEQP + c]);
        float ssum = 0.0f;
        for (int c = 0; c < SEQP; ++c) ssum += __expf(scores[tid * SEQP + c] - m);
        rowm[tid] = m;
        rowinv[tid] = 1.0f / ssum;
    }
    __syncthreads();

    // --- probs -> bf16 in LDS --------------------------------------------
    for (int e = tid; e < 16 * SEQP; e += blockDim.x) {
        int rr = e / SEQP;
        probs[e] = f2bf(__expf(scores[e] - rowm[rr]) * rowinv[rr]);
    }
    __syncthreads();

    // --- ctx: P(16x800) @ V(800x32); 2 N-tiles x 2 K-halves over 4 waves --
    int ntile = wave >> 1, half = wave & 1;
    v8f acc = {};
    for (int kci = half; kci < SEQP / 32; kci += 2) {
        int kk0 = kci * 32;
        v16bf ap = load_a_frag(probs, SEQP, 0, kk0, lane);
        v16bf bv = load_b_frag_strided(vh, HDIM, kk0, ntile * 16, lane);
        acc = wmma_bf16(ap, bv, acc);
    }
    if (half == 1) {
#pragma unroll
        for (int i = 0; i < 8; ++i)
            red[ntile * 256 + (i + hi * 8) * 16 + r] = acc[i];
    }
    __syncthreads();
    if (half == 0) {
#pragma unroll
        for (int i = 0; i < 8; ++i) {
            int row = i + hi * 8;
            float val = acc[i] + red[ntile * 256 + row * 16 + r];
            int s_idx = m0 + row;
            int d_idx = h * HDIM + ntile * 16 + r;
            ctx[((size_t)(b * SEQ + s_idx)) * DMODEL + d_idx] = f2bf(val);
        }
    }
}

// ---------------------------------------------------------------------------
// 8) output projection GEMM: (784x256)@(256x256)^T + bias -> f32
// ---------------------------------------------------------------------------
__global__ void outproj_gemm_kernel(const __bf16* __restrict__ ctx,
                                    const __bf16* __restrict__ wout,
                                    const float* __restrict__ bias,
                                    float* __restrict__ attn_out) {
    int lane = threadIdx.x & 31, wave = threadIdx.x >> 5;
    int tile = blockIdx.x * 8 + wave;
    if (tile >= BATCH * 49 * 16) return;
    int b  = tile / (49 * 16);
    int t  = tile % (49 * 16);
    int m0 = (t / 16) * 16;
    int n0 = (t % 16) * 16;
    const __bf16* Actx = ctx + (size_t)b * SEQ * DMODEL;
    v8f acc = {};
#pragma unroll
    for (int kc = 0; kc < DMODEL; kc += 32) {
        __builtin_prefetch(Actx + (size_t)m0 * DMODEL + kc + 32, 0, 0);
        v16bf a = load_a_frag(Actx, DMODEL, m0, kc, lane);
        v16bf w = load_b_frag_rowk(wout, DMODEL, n0, kc, lane);
        acc = wmma_bf16(a, w, acc);
    }
    int r = lane & 15, hi = lane >> 4;
    float bs = bias[n0 + r];
#pragma unroll
    for (int i = 0; i < 8; ++i) {
        int row = m0 + i + hi * 8;
        attn_out[((size_t)(b * SEQ + row)) * DMODEL + n0 + r] = acc[i] + bs;
    }
}

// ---------------------------------------------------------------------------
// 9) mean pool over sequence -> pooled (32x256)
// ---------------------------------------------------------------------------
__global__ void pool_kernel(const float* __restrict__ attn_out,
                            float* __restrict__ pooled) {
    int idx = blockIdx.x * blockDim.x + threadIdx.x;
    if (idx >= BATCH * DMODEL) return;
    int b = idx >> 8, d = idx & (DMODEL - 1);
    const float* p = attn_out + (size_t)b * SEQ * DMODEL + d;
    float s = 0.0f;
    for (int t = 0; t < SEQ; ++t) s += p[(size_t)t * DMODEL];
    pooled[idx] = s * (1.0f / (float)SEQ);
}

// ---------------------------------------------------------------------------
// host side
// ---------------------------------------------------------------------------
extern "C" void kernel_launch(void* const* d_in, const int* in_sizes, int n_in,
                              void* d_out, int out_size, void* d_ws, size_t ws_size,
                              hipStream_t stream) {
    const float* x      = (const float*)d_in[0];
    const float* conv_w = (const float*)d_in[1];
    const float* CE     = (const float*)d_in[2];
    const float* in_w   = (const float*)d_in[3];
    const float* in_b   = (const float*)d_in[4];
    const float* out_w  = (const float*)d_in[5];
    const float* out_b  = (const float*)d_in[6];
    float* out = (float*)d_out;

    // d_out layout: pooled (32*256) | sparse_weights (32*28*28*64) | top2_idx (32*28*28*2)
    float* pooled_out = out;
    float* sw_out     = out + BATCH * DMODEL;
    float* t2_out     = sw_out + (size_t)NPIX * C1;

    // workspace carve-up (256B aligned)
    char* ws = (char*)d_ws;
    size_t cur = 0;
    auto take = [&](size_t bytes) -> char* {
        char* p = ws + cur;
        cur += (bytes + 255) & ~(size_t)255;
        return p;
    };
    float*    A       = (float*)take((size_t)NCONV * 4);
    unsigned* state   = (unsigned*)take(64);
    unsigned* hist    = (unsigned*)take(4 * 256 * 4);
    float*    pixW    = (float*)take((size_t)NPIX * 2 * 4);
    int*      pixC    = (int*)take((size_t)NPIX * 2 * 4);
    __bf16*   seq     = (__bf16*)take((size_t)NPIX * DMODEL * 2);
    size_t qkv_elems  = (size_t)BATCH * NHEAD * SEQP * HDIM;   // per tensor
    __bf16*   qb      = (__bf16*)take(qkv_elems * 2);
    __bf16*   kb      = (__bf16*)take(qkv_elems * 2);
    __bf16*   vb      = (__bf16*)take(qkv_elems * 2);
    __bf16*   ctx     = (__bf16*)take((size_t)NPIX * DMODEL * 2);
    float*    attn_o  = (float*)take((size_t)NPIX * DMODEL * 4);
    __bf16*   wq_bf   = (__bf16*)take((size_t)3 * DMODEL * DMODEL * 2);
    __bf16*   wo_bf   = (__bf16*)take((size_t)DMODEL * DMODEL * 2);
    (void)ws_size; (void)in_sizes; (void)n_in; (void)out_size;

    // 0) init (zero padded qkv region as one contiguous span + hist + state)
    size_t qkv_words = (3 * qkv_elems * 2) / 4;   // buffers are contiguous
    init_ws_kernel<<<2048, 256, 0, stream>>>((unsigned*)qb, qkv_words, hist, state);

    // 1) conv + relu
    conv_relu_kernel<<<(NCONV + 255) / 256, 256, 0, stream>>>(x, conv_w, A);

    // 2) exact global k-th largest (4 radix passes)
    for (int p = 0; p < 4; ++p) {
        radix_hist_kernel<<<2048, 256, 0, stream>>>(A, NCONV, hist, state, p);
        radix_select_kernel<<<1, 1, 0, stream>>>(state, hist, p);
    }

    // 3) per-pixel sparse softmax (also writes sparse_weights & top2_idx)
    pixel_select_kernel<<<(NPIX + 255) / 256, 256, 0, stream>>>(
        A, state, sw_out, t2_out, pixW, pixC);

    // 4) embedding + positional encoding -> seq bf16
    embed_kernel<<<NPIX, 256, 0, stream>>>(CE, pixW, pixC, seq);

    // 5) weights -> bf16
    convert_w_kernel<<<(3 * DMODEL * DMODEL + 255) / 256, 256, 0, stream>>>(
        in_w, out_w, wq_bf, wo_bf);

    // 6) QKV projection (WMMA)
    qkv_gemm_kernel<<<(BATCH * 49 * 48) / 8, 256, 0, stream>>>(
        seq, wq_bf, in_b, qb, kb, vb);

    // 7) fused attention (WMMA scores + softmax + WMMA P@V)
    size_t smem = (16 * SEQP + 16 + 16 + 2 * 256) * sizeof(float)
                + (size_t)16 * SEQP * sizeof(__bf16);   // 78,976 B
    attn_kernel<<<BATCH * NHEAD * 49, 128, smem, stream>>>(qb, kb, vb, ctx);

    // 8) output projection (WMMA)
    outproj_gemm_kernel<<<(BATCH * 49 * 16) / 8, 256, 0, stream>>>(
        ctx, wo_bf, out_b, attn_o);

    // 9) mean pool
    pool_kernel<<<(BATCH * DMODEL + 255) / 256, 256, 0, stream>>>(attn_o, pooled_out);
}
